// BaselineDenseMamba_64484638982269
// MI455X (gfx1250) — matrix operand order repaired and checked by reading, not compiled, not run on hardware
//
#include <hip/hip_runtime.h>
#include <hip/hip_bf16.h>

// ---------------------------------------------------------------------------
// Mamba block for MI455X (gfx1250, wave32, WMMA).
//   1) f32->bf16 converts (x_proj_w zero-padded 96->128 rows so ALL GEMMs are
//      exact multiples of the 128x64x32 block tile -> guard-free inner loop)
//   2) WMMA bf16 GEMM: xz = hs @ in_proj_w^T          (8192x1024x4096)
//   3) depthwise conv(d_conv=4) + SiLU
//   4) WMMA GEMM: x_dbl = xconv @ x_proj_w^T          (8192x2048x128pad)
//   5) slice dt_rank cols -> bf16
//   6) WMMA GEMM: dt_raw = dt @ dt_proj_w^T           (8192x64x2048)
//   7) selective scan (softplus+bias, D skip, SiLU(z) gate) -> y bf16
//   8) WMMA GEMM: out = y @ out_proj_w^T -> d_out f32 (8192x2048x1024)
// GEMM inner loop: register-staged pipeline; globals for chunk k+1 in flight
// under the v_wmma's for chunk k; prefetch chunk k+2.
// ---------------------------------------------------------------------------

typedef __attribute__((ext_vector_type(16))) __bf16 v16bf;
typedef __attribute__((ext_vector_type(8)))  float  v8f;

__device__ __forceinline__ unsigned short f32_to_bf16_rne(float f) {
    unsigned int u = __float_as_uint(f);
    unsigned int r = u + 0x7FFFu + ((u >> 16) & 1u);
    return (unsigned short)(r >> 16);
}

__device__ __forceinline__ float silu(float x) {
    return x / (1.0f + __expf(-x));
}

// ---------------------------------------------------------------------------
// f32 -> bf16 conversion with zero padding beyond n_src (up to n_total)
// ---------------------------------------------------------------------------
__global__ void __launch_bounds__(256)
cvt_f32_bf16_kernel(const float* __restrict__ src, unsigned short* __restrict__ dst,
                    size_t n_src, size_t n_total) {
    size_t i = (size_t)blockIdx.x * blockDim.x + threadIdx.x;
    if (i >= n_total) return;
    dst[i] = (i < n_src) ? f32_to_bf16_rne(src[i]) : (unsigned short)0;
}

// ---------------------------------------------------------------------------
// WMMA bf16 GEMM:  C[M,N] (f32) = A[M,K] (bf16) * W[N,K]^T (bf16)
// Preconditions: M % 128 == 0, N % 64 == 0, K % 32 == 0 (caller guarantees).
// Block tile 128(M) x 64(N), K-step 32, 256 threads = 8 wave32 waves.
// Wave (wm,wn) in 4x2 grid owns a 32x32 sub-tile = 2x2 WMMA 16x16 frags.
// ---------------------------------------------------------------------------
#define GEMM_BM 128
#define GEMM_BN 64
#define GEMM_BK 32
#define GEMM_LD 40   // padded LDS row stride (ushorts); 80B keeps 16B groups aligned

union FragBF16 {
    uint4 q[2];
    v16bf v;
};

__global__ void __launch_bounds__(256)
gemm_bf16_wmma_kernel(const unsigned short* __restrict__ A,   // [M,K] bf16 row-major
                      const unsigned short* __restrict__ W,   // [N,K] bf16 row-major
                      float* __restrict__ C,                  // [M,N] f32
                      int M, int N, int K) {
    (void)M;
    __shared__ __align__(16) unsigned short As[GEMM_BM * GEMM_LD];
    __shared__ __align__(16) unsigned short Ws[GEMM_BN * GEMM_LD];

    const int tid  = threadIdx.x;
    const int lane = tid & 31;
    const int wave = tid >> 5;
    const int wm   = wave & 3;   // 0..3 -> M sub-tile
    const int wn   = wave >> 2;  // 0..1 -> N sub-tile
    const int lo   = lane & 15;
    const int hi   = lane >> 4;

    const int m0 = blockIdx.y * GEMM_BM;
    const int n0 = blockIdx.x * GEMM_BN;

    // A loader: thread -> (row r = tid/2, half h = tid%2): 2x uint4 = 16 bf16
    const int ar = tid >> 1;
    const int ah = tid & 1;
    // W loader: thread -> (row rn = tid/4, quarter q = tid%4): 1x uint4 = 8 bf16
    const int wr = tid >> 2;
    const int wq = tid & 3;

    const unsigned short* aptr = A + (size_t)(m0 + ar) * K + ah * 16;
    const unsigned short* wptr = W + (size_t)(n0 + wr) * K + wq * 8;

    v8f acc[2][2];
    #pragma unroll
    for (int i = 0; i < 2; ++i)
        #pragma unroll
        for (int j = 0; j < 2; ++j)
            acc[i][j] = (v8f)0.0f;

    // preload chunk 0 into registers
    uint4 areg0 = ((const uint4*)aptr)[0];
    uint4 areg1 = ((const uint4*)aptr)[1];
    uint4 wreg  = ((const uint4*)wptr)[0];
    aptr += GEMM_BK;
    wptr += GEMM_BK;

    for (int kb = 0; kb < K; kb += GEMM_BK) {
        // ---- commit staged registers to LDS ----
        {
            uint4* da = (uint4*)&As[ar * GEMM_LD + ah * 16];
            da[0] = areg0;
            da[1] = areg1;
            *(uint4*)&Ws[wr * GEMM_LD + wq * 8] = wreg;
        }
        __syncthreads();

        // ---- issue global loads for chunk k+1 (in flight under the WMMAs) ----
        if (kb + GEMM_BK < K) {
            areg0 = ((const uint4*)aptr)[0];
            areg1 = ((const uint4*)aptr)[1];
            wreg  = ((const uint4*)wptr)[0];
            // prefetch chunk k+2 (global_prefetch_b8)
            __builtin_prefetch(aptr + GEMM_BK, 0, 1);
            __builtin_prefetch(wptr + GEMM_BK, 0, 1);
            aptr += GEMM_BK;
            wptr += GEMM_BK;
        }

        // ---- fragments from LDS ----
        // A frag (16x32): lane half selects K groups {0-7,16-23} vs {8-15,24-31}
        FragBF16 af[2], bf[2];
        #pragma unroll
        for (int i = 0; i < 2; ++i) {
            const int m = wm * 32 + i * 16 + lo;
            af[i].q[0] = *(const uint4*)&As[m * GEMM_LD + 8 * hi];
            af[i].q[1] = *(const uint4*)&As[m * GEMM_LD + 16 + 8 * hi];
        }
        // B frag (32x16): lane = column n, K groups {0-7,16-23} vs {8-15,24-31} by lane half
        #pragma unroll
        for (int j = 0; j < 2; ++j) {
            const int n = wn * 32 + j * 16 + lo;
            bf[j].q[0] = *(const uint4*)&Ws[n * GEMM_LD + 8 * hi];
            bf[j].q[1] = *(const uint4*)&Ws[n * GEMM_LD + 16 + 8 * hi];
        }

        #pragma unroll
        for (int i = 0; i < 2; ++i)
            #pragma unroll
            for (int j = 0; j < 2; ++j)
                acc[i][j] = __builtin_amdgcn_wmma_f32_16x16x32_bf16(
                    false, af[i].v, false, bf[j].v,
                    (short)0, acc[i][j], false, false);

        __syncthreads();
    }

    // ---- store C: vgpr r, lanes0-15 -> M=r, lanes16-31 -> M=8+r; N = lane&15 ----
    #pragma unroll
    for (int i = 0; i < 2; ++i) {
        #pragma unroll
        for (int j = 0; j < 2; ++j) {
            const int ng = n0 + wn * 32 + j * 16 + lo;
            #pragma unroll
            for (int r = 0; r < 8; ++r) {
                const int mg = m0 + wm * 32 + i * 16 + 8 * hi + r;
                C[(size_t)mg * N + ng] = acc[i][j][r];
            }
        }
    }
}

// ---------------------------------------------------------------------------
// Depthwise causal conv (d_conv=4) + bias + SiLU. Reads x-half of xz.
// xz layout: [B*L, 4096], x = cols [0,2048).
// ---------------------------------------------------------------------------
__global__ void __launch_bounds__(256)
conv_silu_kernel(const float* __restrict__ xz,
                 const float* __restrict__ conv_w,  // [2048,4]
                 const float* __restrict__ conv_b,  // [2048]
                 float* __restrict__ xconv_f32,     // [B*L,2048]
                 unsigned short* __restrict__ xconv_bf16,
                 int L) {
    const size_t idx = (size_t)blockIdx.x * blockDim.x + threadIdx.x;
    const int d  = (int)(idx & 2047);
    const size_t bt = idx >> 11;
    const int t = (int)(bt % (size_t)L);
    const size_t brow = bt - (size_t)t;   // b*L

    float s = conv_b[d];
    #pragma unroll
    for (int j = 0; j < 4; ++j) {
        const int tt = t - 3 + j;
        if (tt >= 0)
            s += conv_w[d * 4 + j] * xz[(brow + (size_t)tt) * 4096 + d];
    }
    const float v = silu(s);
    xconv_f32[idx]  = v;
    xconv_bf16[idx] = f32_to_bf16_rne(v);
}

// ---------------------------------------------------------------------------
// Slice dt_rank columns of x_dbl [M,128pad] -> bf16 [M,64]
// ---------------------------------------------------------------------------
__global__ void __launch_bounds__(256)
slice_dt_kernel(const float* __restrict__ x_dbl, unsigned short* __restrict__ dt_bf16, size_t n) {
    size_t i = (size_t)blockIdx.x * blockDim.x + threadIdx.x;
    if (i >= n) return;
    size_t row = i >> 6;
    int    r   = (int)(i & 63);
    dt_bf16[i] = f32_to_bf16_rne(x_dbl[row * 128 + r]);
}

// ---------------------------------------------------------------------------
// Selective scan. One thread per (b, d) channel; 16 states in registers.
// Fuses: softplus(dt + bias), recurrence, C contraction, D skip, SiLU(z) gate.
// ---------------------------------------------------------------------------
__global__ void __launch_bounds__(256)
scan_kernel(const float* __restrict__ dt_raw,   // [B*L,2048]
            const float* __restrict__ xconv,    // [B*L,2048]
            const float* __restrict__ xz,       // [B*L,4096] (z = cols 2048..4095)
            const float* __restrict__ x_dbl,    // [B*L,128pad] (B at 64, C at 80)
            const float* __restrict__ dt_b,     // [2048]
            const float* __restrict__ A_log,    // [2048,16]
            const float* __restrict__ Dvec,     // [2048]
            unsigned short* __restrict__ y_bf16,
            int L) {
    const int gid = blockIdx.x * blockDim.x + threadIdx.x;  // 0..4095
    const int b = gid >> 11;
    const int d = gid & 2047;

    float Arow[16];
    #pragma unroll
    for (int n = 0; n < 16; ++n) Arow[n] = -__expf(A_log[d * 16 + n]);

    float h[16];
    #pragma unroll
    for (int n = 0; n < 16; ++n) h[n] = 0.0f;

    const float Dd  = Dvec[d];
    const float dtb = dt_b[d];
    const size_t base = (size_t)b * (size_t)L;

    for (int t = 0; t < L; ++t) {
        const size_t row = base + (size_t)t;
        float dtv = dt_raw[row * 2048 + d] + dtb;
        dtv = (dtv > 20.0f) ? dtv : log1pf(__expf(dtv));   // softplus
        const float xv = xconv[row * 2048 + d];
        const float zv = xz[row * 4096 + 2048 + d];
        const float dtx = dtv * xv;
        const float* bc = x_dbl + row * 128;

        float y = 0.0f;
        #pragma unroll
        for (int n = 0; n < 16; ++n) {
            const float dA = __expf(dtv * Arow[n]);
            h[n] = h[n] * dA + dtx * bc[64 + n];
            y += h[n] * bc[80 + n];
        }
        y += xv * Dd;
        y *= silu(zv);
        y_bf16[row * 2048 + d] = f32_to_bf16_rne(y);
    }
}

// ---------------------------------------------------------------------------
// Host-side orchestration
// ---------------------------------------------------------------------------
static inline void* ws_alloc(char* base, size_t& cur, size_t bytes) {
    void* p = base + cur;
    cur += (bytes + 255) & ~(size_t)255;
    return p;
}

extern "C" void kernel_launch(void* const* d_in, const int* in_sizes, int n_in,
                              void* d_out, int out_size, void* d_ws, size_t ws_size,
                              hipStream_t stream) {
    (void)in_sizes; (void)n_in; (void)out_size; (void)ws_size;

    const int Bb = 2, L = 4096, d_model = 1024;
    const int d_inner = 2048, dt_rank = 64;
    const int M = Bb * L;                 // 8192 rows
    const int E = 2 * d_inner;            // 4096
    const int NK = dt_rank + 2 * 16;      // 96 (real x_proj rows)
    const int NKp = 128;                  // padded to a multiple of GEMM_BN

    const float* hs        = (const float*)d_in[0];
    const float* in_proj_w = (const float*)d_in[1];
    const float* conv_w    = (const float*)d_in[2];
    const float* conv_b    = (const float*)d_in[3];
    const float* x_proj_w  = (const float*)d_in[4];
    const float* dt_proj_w = (const float*)d_in[5];
    const float* dt_proj_b = (const float*)d_in[6];
    const float* A_log     = (const float*)d_in[7];
    const float* Dvec      = (const float*)d_in[8];
    const float* out_proj_w= (const float*)d_in[9];
    float* out = (float*)d_out;

    char* ws = (char*)d_ws;
    size_t cur = 0;
    unsigned short* hs_bf    = (unsigned short*)ws_alloc(ws, cur, (size_t)M * d_model * 2);
    unsigned short* win_bf   = (unsigned short*)ws_alloc(ws, cur, (size_t)E * d_model * 2);
    float*          xz       = (float*)         ws_alloc(ws, cur, (size_t)M * E * 4);
    float*          xconv    = (float*)         ws_alloc(ws, cur, (size_t)M * d_inner * 4);
    unsigned short* xconv_bf = (unsigned short*)ws_alloc(ws, cur, (size_t)M * d_inner * 2);
    unsigned short* wxp_bf   = (unsigned short*)ws_alloc(ws, cur, (size_t)NKp * d_inner * 2);
    float*          x_dbl    = (float*)         ws_alloc(ws, cur, (size_t)M * NKp * 4);
    unsigned short* dtin_bf  = (unsigned short*)ws_alloc(ws, cur, (size_t)M * dt_rank * 2);
    unsigned short* wdt_bf   = (unsigned short*)ws_alloc(ws, cur, (size_t)d_inner * dt_rank * 2);
    float*          dt_raw   = (float*)         ws_alloc(ws, cur, (size_t)M * d_inner * 4);
    unsigned short* y_bf     = (unsigned short*)ws_alloc(ws, cur, (size_t)M * d_inner * 2);
    unsigned short* wout_bf  = (unsigned short*)ws_alloc(ws, cur, (size_t)d_model * d_inner * 2);

    auto cvt = [&](const float* src, unsigned short* dst, size_t n_src, size_t n_total) {
        cvt_f32_bf16_kernel<<<(unsigned)((n_total + 255) / 256), 256, 0, stream>>>(
            src, dst, n_src, n_total);
    };

    // 1) precision converts (x_proj_w zero-padded to 128 rows)
    cvt(hs,         hs_bf,   (size_t)M * d_model,        (size_t)M * d_model);
    cvt(in_proj_w,  win_bf,  (size_t)E * d_model,        (size_t)E * d_model);
    cvt(x_proj_w,   wxp_bf,  (size_t)NK * d_inner,       (size_t)NKp * d_inner);
    cvt(dt_proj_w,  wdt_bf,  (size_t)d_inner * dt_rank,  (size_t)d_inner * dt_rank);
    cvt(out_proj_w, wout_bf, (size_t)d_model * d_inner,  (size_t)d_model * d_inner);

    // 2) xz = hs @ in_proj_w^T   [8192 x 4096], K=1024
    {
        dim3 grid(E / GEMM_BN, M / GEMM_BM);
        gemm_bf16_wmma_kernel<<<grid, 256, 0, stream>>>(hs_bf, win_bf, xz, M, E, d_model);
    }

    // 3) depthwise conv + SiLU
    {
        size_t total = (size_t)M * d_inner;
        conv_silu_kernel<<<(unsigned)(total / 256), 256, 0, stream>>>(
            xz, conv_w, conv_b, xconv, xconv_bf, L);
    }

    // 4) x_dbl = xconv @ x_proj_w^T   [8192 x 128pad], K=2048
    {
        dim3 grid(NKp / GEMM_BN, M / GEMM_BM);
        gemm_bf16_wmma_kernel<<<grid, 256, 0, stream>>>(xconv_bf, wxp_bf, x_dbl, M, NKp, d_inner);
    }

    // 5) dt slice -> bf16
    {
        size_t total = (size_t)M * dt_rank;
        slice_dt_kernel<<<(unsigned)((total + 255) / 256), 256, 0, stream>>>(x_dbl, dtin_bf, total);
    }

    // 6) dt_raw = dt @ dt_proj_w^T   [8192 x 2048], K=64
    {
        dim3 grid(d_inner / GEMM_BN, M / GEMM_BM);
        gemm_bf16_wmma_kernel<<<grid, 256, 0, stream>>>(dtin_bf, wdt_bf, dt_raw, M, d_inner, dt_rank);
    }

    // 7) selective scan (fused softplus, D skip, SiLU(z) gate) -> y bf16
    {
        scan_kernel<<<(Bb * d_inner) / 256, 256, 0, stream>>>(
            dt_raw, xconv, xz, x_dbl, dt_proj_b, A_log, Dvec, y_bf, L);
    }

    // 8) out = y @ out_proj_w^T -> d_out   [8192 x 1024], K=2048
    {
        dim3 grid(d_model / GEMM_BN, M / GEMM_BM);
        gemm_bf16_wmma_kernel<<<grid, 256, 0, stream>>>(y_bf, wout_bf, out, M, d_model, d_inner);
    }
}
